// SS2DGlobal_11819749998668
// MI455X (gfx1250) — compile-verified, matrix operand order checked by reading, not run on hardware
//
#include <hip/hip_runtime.h>
#include <hip/hip_bf16.h>

// ---------------- problem constants (from setup_inputs) ----------------
#define BB   8
#define HH   48
#define WW2  48
#define DM   96     // d_model
#define DIN  192    // d_inner
#define NS   16     // state size N
#define RR   6      // dt rank
#define KK   4      // scan directions
#define LL   (HH*WW2)       // 2304
#define BL   (BB*LL)        // 18432
#define TT   64             // scan timesteps per TDM tile
#define NT   (LL/TT)        // 36 tiles

typedef __attribute__((ext_vector_type(16))) _Float16 v16h;
typedef __attribute__((ext_vector_type(8)))  float    v8f;
typedef __attribute__((ext_vector_type(4)))  unsigned int u32x4;
typedef __attribute__((ext_vector_type(8)))  int      i32x8;
typedef __attribute__((ext_vector_type(4)))  int      i32x4;

__device__ __forceinline__ float sigmoidf_(float x) {
  // v_exp_f32 + v_add + v_rcp_f32 (avoid IEEE divide expansion)
  return __builtin_amdgcn_rcpf(1.f + __expf(-x));
}

// ---------------- prep: f32 -> f16 conversion ----------------
__global__ void cvt_f16_kernel(const float* __restrict__ s, _Float16* __restrict__ d, int n) {
  int i = blockIdx.x * blockDim.x + threadIdx.x;
  if (i < n) d[i] = (_Float16)s[i];
}

// pad x_proj_w (K,38,DIN) -> f16 (K,48,DIN), rows 38..47 zero
__global__ void pad_xprojw_kernel(const float* __restrict__ s, _Float16* __restrict__ d) {
  int i = blockIdx.x * blockDim.x + threadIdx.x;
  if (i >= KK * 48 * DIN) return;
  int dd = i % DIN; int c = (i / DIN) % 48; int k = i / (48 * DIN);
  float v = (c < RR + 2 * NS) ? s[(k * (RR + 2 * NS) + c) * DIN + dd] : 0.f;
  d[i] = (_Float16)v;
}

// pad dt_projs_w (K,DIN,6) -> f16 (K,DIN,32), cols 6..31 zero
__global__ void pad_dtw_kernel(const float* __restrict__ s, _Float16* __restrict__ d) {
  int i = blockIdx.x * blockDim.x + threadIdx.x;
  if (i >= KK * DIN * 32) return;
  int r = i & 31; int dd = (i >> 5) % DIN; int k = i / (32 * DIN);
  float v = (r < RR) ? s[(k * DIN + dd) * RR + r] : 0.f;
  d[i] = (_Float16)v;
}

// ---------------- WMMA inner loop ----------------
__device__ __forceinline__ v8f wmma_loop(const _Float16* __restrict__ ap,
                                         const _Float16* __restrict__ bp, int kdim) {
  v8f acc = {};
  for (int k = 0; k < kdim; k += 32) {
    v16h a = *(const v16h*)(ap + k);
    v16h b = *(const v16h*)(bp + k);
    acc = __builtin_amdgcn_wmma_f32_16x16x32_f16(false, a, false, b, (short)0, acc, false, false);
  }
  return acc;
}

// ---------------- in_proj GEMM: xz = x @ W^T, split -> xc (NCHW) / z=silu ----------------
__global__ void inproj_gemm_kernel(const _Float16* __restrict__ Ah, const _Float16* __restrict__ Wh,
                                   float* __restrict__ xc, float* __restrict__ z) {
  int lane = threadIdx.x & 31, wave = threadIdx.x >> 5;
  int r = lane & 15, half = lane >> 4;
  int m0 = blockIdx.x * 16;
  int nt = blockIdx.y * 4 + wave;        // 0..23
  int n = nt * 16 + r;
  const _Float16* ap = Ah + (size_t)(m0 + r) * DM + half * 16;
  const _Float16* bp = Wh + (size_t)n * DM + half * 16;
  v8f acc = wmma_loop(ap, bp, DM);
  int b = m0 / LL;
  int lbase = m0 - b * LL;
#pragma unroll
  for (int v = 0; v < 8; ++v) {
    int l = lbase + v + half * 8;
    float val = acc[v];
    if (n < DIN) {
      xc[((size_t)(b * DIN + n)) * LL + l] = val;           // (B,Din,H,W) for conv
    } else {
      z[((size_t)(b * LL + l)) * DIN + (n - DIN)] = val * sigmoidf_(val);
    }
  }
}

// ---------------- depthwise 3x3 SAME conv + SiLU + cross-scan (4 dirs) ----------------
__global__ void conv_scan_kernel(const float* __restrict__ xc, const float* __restrict__ cw,
                                 const float* __restrict__ cb, _Float16* __restrict__ xsT) {
  int i = blockIdx.x * blockDim.x + threadIdx.x;
  if (i >= BB * DIN * LL) return;
  int l = i % LL; int d = (i / LL) % DIN; int b = i / (LL * DIN);
  int h = l / WW2, w = l % WW2;
  const float* plane = xc + ((size_t)(b * DIN + d)) * LL;
  float acc = cb[d];
#pragma unroll
  for (int ky = 0; ky < 3; ++ky) {
    int hh = h + ky - 1;
    if (hh < 0 || hh >= HH) continue;
#pragma unroll
    for (int kx = 0; kx < 3; ++kx) {
      int ww = w + kx - 1;
      if (ww < 0 || ww >= WW2) continue;
      acc += plane[hh * WW2 + ww] * cw[d * 9 + ky * 3 + kx];
    }
  }
  acc = acc * sigmoidf_(acc);
  _Float16 hv = (_Float16)acc;
  int l1 = w * HH + h;
  size_t base = (size_t)(b * KK) * LL * DIN;
  xsT[base + ((size_t)(0 * LL + l)) * DIN + d] = hv;
  xsT[base + ((size_t)(1 * LL + l1)) * DIN + d] = hv;
  xsT[base + ((size_t)(2 * LL + (LL - 1 - l))) * DIN + d] = hv;
  xsT[base + ((size_t)(3 * LL + (LL - 1 - l1))) * DIN + d] = hv;
}

// ---------------- x_proj GEMM per (b,k) ----------------
__global__ void xproj_gemm_kernel(const _Float16* __restrict__ xsT, const _Float16* __restrict__ Wh,
                                  _Float16* __restrict__ dtsR, float* __restrict__ BC) {
  int lane = threadIdx.x & 31, wave = threadIdx.x >> 5;
  int r = lane & 15, half = lane >> 4;
  int bk = blockIdx.z; int k = bk & 3;
  int m0 = blockIdx.x * 16;
  int c = wave * 16 + r;                 // 0..47
  const _Float16* ap = xsT + ((size_t)bk * LL + m0 + r) * DIN + half * 16;
  const _Float16* bp = Wh + ((size_t)(k * 48 + c)) * DIN + half * 16;
  v8f acc = wmma_loop(ap, bp, DIN);
#pragma unroll
  for (int v = 0; v < 8; ++v) {
    int l = m0 + v + half * 8;
    float val = acc[v];
    size_t row = ((size_t)bk * LL + l) * 32;
    if (c < RR)        dtsR[row + c] = (_Float16)val;
    else if (c < 32)   dtsR[row + c] = (_Float16)0.f;
    if (c >= RR && c < RR + 2 * NS) BC[row + (c - RR)] = val;
  }
}

// ---------------- dt GEMM per (b,k) ----------------
__global__ void dt_gemm_kernel(const _Float16* __restrict__ dtsR, const _Float16* __restrict__ Wh,
                               _Float16* __restrict__ dtT) {
  int lane = threadIdx.x & 31, wave = threadIdx.x >> 5;
  int r = lane & 15, half = lane >> 4;
  int bk = blockIdx.z; int k = bk & 3;
  int m0 = blockIdx.x * 16;
  int n = (blockIdx.y * 4 + wave) * 16 + r;
  const _Float16* ap = dtsR + ((size_t)bk * LL + m0 + r) * 32 + half * 16;
  const _Float16* bp = Wh + ((size_t)(k * DIN + n)) * 32 + half * 16;
  v8f acc = wmma_loop(ap, bp, 32);
#pragma unroll
  for (int v = 0; v < 8; ++v) {
    int l = m0 + v + half * 8;
    dtT[((size_t)bk * LL + l) * DIN + n] = (_Float16)acc[v];
  }
}

// ---------------- Tensor Data Mover: stage tile into LDS ----------------
// D# groups per CDNA5 ISA ch.8: group0 = {count/flags, lds_addr, global_addr lo/hi+type},
// group1 = {data_size, tensor dims (huge -> no OOB), tile dims, dim0 stride}.
// This toolchain's builtin is the 6-arg form: (g0, g1, g2, g3, g4, cpol).
__device__ __forceinline__ void tdm_load(unsigned lds_off, unsigned long long ga,
                                         int dsz_log, int tile_d0, int tile_d1, int stride0) {
  u32x4 g0;
  g0[0] = 1u;                                    // count=1, user mode
  g0[1] = lds_off;                               // LDS byte address
  g0[2] = (unsigned)ga;                          // global_addr[31:0]
  g0[3] = (unsigned)(ga >> 32) | (2u << 30);     // global_addr[56:32] | type=2
  i32x8 g1 = (i32x8)0;
  g1[0] = dsz_log << 16;                         // workgroup_mask=0, data_size
  g1[1] = 0;                                     // tensor_dim0 = 1<<30 (lo16=0)
  g1[2] = 0x4000;                                // tensor_dim0 hi16; tensor_dim1 lo16=0
  g1[3] = 0x4000 | (tile_d0 << 16);              // tensor_dim1 hi16 (1<<30); tile_dim0
  g1[4] = tile_d1 & 0xFFFF;                      // tile_dim1 (0 => 1D); tile_dim2=0
  g1[5] = stride0;                               // tensor_dim0_stride[31:0]
  i32x4 gz4 = (i32x4)0;
  i32x8 gz8 = (i32x8)0;
  __builtin_amdgcn_tensor_load_to_lds(g0, g1, gz4, gz4, gz8, 0);
}

__device__ __forceinline__ unsigned lds_off_of(const void* p) {
  // flat LDS aperture keeps the LDS byte offset in addr[31:0]
  return (unsigned)(unsigned long long)p;
}

// ---------------- selective scan with TDM double-buffered staging ----------------
// block = 256 threads = one (b,k), 16 d-channels x 16 states; lane = (d_local, n)
__global__ void scan_kernel(const _Float16* __restrict__ xsT, const _Float16* __restrict__ dtT,
                            const float* __restrict__ BC, const float* __restrict__ A_logs,
                            const float* __restrict__ Ds, const float* __restrict__ dtb,
                            float* __restrict__ ys) {
  __shared__ float    bcbuf[2][TT * 32];   // 16 KB
  __shared__ _Float16 ubuf [2][TT * 16];   // 4 KB
  __shared__ _Float16 dbuf [2][TT * 16];   // 4 KB

  int chan0 = blockIdx.x * 16;             // first (b,k,d) channel of this block
  int bk = chan0 / DIN;
  int d0 = chan0 % DIN;
  int k = bk & 3;
  int dl = threadIdx.x >> 4;               // 0..15 local channel
  int n  = threadIdx.x & 15;               // state index
  int d  = d0 + dl;

  const _Float16* up = xsT + (size_t)bk * LL * DIN + d0;
  const _Float16* dp = dtT + (size_t)bk * LL * DIN + d0;
  const float* bcp = BC + (size_t)bk * LL * 32;
  float* yp = ys + (size_t)bk * LL * DIN + d;

  float Aval = -__expf(A_logs[(k * DIN + d) * NS + n]);
  float Dval = Ds[k * DIN + d];
  float bias = dtb[k * DIN + d];
  float hst = 0.f;

  bool issuer = (threadIdx.x >> 5) == 0;   // wave 0 drives the TDM
  if (issuer) {
    tdm_load(lds_off_of(&bcbuf[0][0]), (unsigned long long)(bcp), 2, TT * 32, 0, TT * 32);
    tdm_load(lds_off_of(&ubuf[0][0]),  (unsigned long long)(up),  1, 16, TT, DIN);
    tdm_load(lds_off_of(&dbuf[0][0]),  (unsigned long long)(dp),  1, 16, TT, DIN);
  }

  for (int tile = 0; tile < NT; ++tile) {
    int cur = tile & 1;
    if (issuer) {
      if (tile + 1 < NT) {
        int nx = (tile + 1) & 1;
        const float* bcn = bcp + (size_t)(tile + 1) * TT * 32;
        const _Float16* un = up + (size_t)(tile + 1) * TT * DIN;
        const _Float16* dn = dp + (size_t)(tile + 1) * TT * DIN;
        tdm_load(lds_off_of(&bcbuf[nx][0]), (unsigned long long)bcn, 2, TT * 32, 0, TT * 32);
        tdm_load(lds_off_of(&ubuf[nx][0]),  (unsigned long long)un,  1, 16, TT, DIN);
        tdm_load(lds_off_of(&dbuf[nx][0]),  (unsigned long long)dn,  1, 16, TT, DIN);
        __builtin_amdgcn_s_wait_tensorcnt((short)3);   // current tile's 3 loads done (in-order)
      } else {
        __builtin_amdgcn_s_wait_tensorcnt((short)0);
      }
    }
    __syncthreads();                       // tile `cur` resident for all waves

    for (int tt = 0; tt < TT; ++tt) {
      float u = (float)ubuf[cur][tt * 16 + dl];
      float x = (float)dbuf[cur][tt * 16 + dl] + bias;
      float delta = (x > 20.f) ? x : __logf(1.f + __expf(x));   // softplus
      float Bv = bcbuf[cur][tt * 32 + n];
      float Cv = bcbuf[cur][tt * 32 + 16 + n];
      hst = __expf(delta * Aval) * hst + (delta * u) * Bv;
      float p = hst * Cv;
      p += __shfl_xor(p, 1, 16);
      p += __shfl_xor(p, 2, 16);
      p += __shfl_xor(p, 4, 16);
      p += __shfl_xor(p, 8, 16);
      if (n == 0) yp[(size_t)(tile * TT + tt) * DIN] = p + Dval * u;
    }
    __syncthreads();                       // all readers done before buffer is reloaded
  }
}

// ---------------- cross-merge + LayerNorm + gate ----------------
__global__ void merge_ln_gate_kernel(const float* __restrict__ ys, const float* __restrict__ z,
                                     const float* __restrict__ g, const float* __restrict__ bbias,
                                     _Float16* __restrict__ yg) {
  int bl = blockIdx.x; int b = bl / LL; int l = bl % LL;
  int h = l / WW2, w = l % WW2;
  int lt = w * HH + h;
  int d = threadIdx.x;                    // 0..191
  size_t base = (size_t)(b * KK) * LL * DIN;
  float y = ys[base + ((size_t)(0 * LL + l)) * DIN + d]
          + ys[base + ((size_t)(2 * LL + (LL - 1 - l))) * DIN + d]
          + ys[base + ((size_t)(1 * LL + lt)) * DIN + d]
          + ys[base + ((size_t)(3 * LL + (LL - 1 - lt))) * DIN + d];
  float s = y, s2 = y * y;
#pragma unroll
  for (int off = 1; off < 32; off <<= 1) {
    s += __shfl_xor(s, off, 32);
    s2 += __shfl_xor(s2, off, 32);
  }
  __shared__ float red[12];
  int wave = threadIdx.x >> 5, lane = threadIdx.x & 31;
  if (lane == 0) { red[wave] = s; red[6 + wave] = s2; }
  __syncthreads();
  float ts = 0.f, ts2 = 0.f;
#pragma unroll
  for (int i = 0; i < 6; ++i) { ts += red[i]; ts2 += red[6 + i]; }
  float mu = ts * (1.f / DIN);
  float var = ts2 * (1.f / DIN) - mu * mu;
  float yl = (y - mu) * rsqrtf(var + 1e-5f) * g[d] + bbias[d];
  float zz = z[((size_t)bl) * DIN + d];
  yg[((size_t)bl) * DIN + d] = (_Float16)(yl * zz);
}

// ---------------- out_proj GEMM ----------------
__global__ void out_gemm_kernel(const _Float16* __restrict__ yg, const _Float16* __restrict__ Wh,
                                float* __restrict__ out) {
  int lane = threadIdx.x & 31, wave = threadIdx.x >> 5;
  int r = lane & 15, half = lane >> 4;
  int m0 = blockIdx.x * 16;
  int n = (blockIdx.y * 3 + wave) * 16 + r;
  const _Float16* ap = yg + (size_t)(m0 + r) * DIN + half * 16;
  const _Float16* bp = Wh + (size_t)n * DIN + half * 16;
  v8f acc = wmma_loop(ap, bp, DIN);
#pragma unroll
  for (int v = 0; v < 8; ++v) {
    int m = m0 + v + half * 8;
    out[(size_t)m * DM + n] = acc[v];
  }
}

// ---------------- launcher ----------------
extern "C" void kernel_launch(void* const* d_in, const int* in_sizes, int n_in,
                              void* d_out, int out_size, void* d_ws, size_t ws_size,
                              hipStream_t stream) {
  (void)in_sizes; (void)n_in; (void)out_size; (void)ws_size;
  const float* x       = (const float*)d_in[0];
  const float* w_in    = (const float*)d_in[1];
  const float* conv_w  = (const float*)d_in[2];
  const float* conv_b  = (const float*)d_in[3];
  const float* xproj_w = (const float*)d_in[4];
  const float* dt_w    = (const float*)d_in[5];
  const float* dt_b    = (const float*)d_in[6];
  const float* A_logs  = (const float*)d_in[7];
  const float* Ds      = (const float*)d_in[8];
  const float* ln_g    = (const float*)d_in[9];
  const float* ln_b    = (const float*)d_in[10];
  const float* w_out   = (const float*)d_in[11];
  float* out = (float*)d_out;

  char* ws = (char*)d_ws;
  size_t off = 0;
  auto alloc = [&](size_t bytes) -> char* {
    char* p = ws + off;
    off += (bytes + 255) & ~(size_t)255;
    return p;
  };
  _Float16* x_h    = (_Float16*)alloc((size_t)BL * DM * 2);
  _Float16* win_h  = (_Float16*)alloc((size_t)2 * DIN * DM * 2);
  _Float16* wout_h = (_Float16*)alloc((size_t)DM * DIN * 2);
  _Float16* xpw_h  = (_Float16*)alloc((size_t)KK * 48 * DIN * 2);
  _Float16* dtw_h  = (_Float16*)alloc((size_t)KK * DIN * 32 * 2);
  float*    xc     = (float*)   alloc((size_t)BB * DIN * LL * 4);
  float*    zb     = (float*)   alloc((size_t)BL * DIN * 4);
  _Float16* xsT    = (_Float16*)alloc((size_t)BB * KK * LL * DIN * 2);
  _Float16* dtsR   = (_Float16*)alloc((size_t)BB * KK * LL * 32 * 2);
  float*    BCb    = (float*)   alloc((size_t)BB * KK * LL * 32 * 4);
  _Float16* dtT    = (_Float16*)alloc((size_t)BB * KK * LL * DIN * 2);
  float*    ysb    = (float*)   alloc((size_t)BB * KK * LL * DIN * 4);
  _Float16* yg     = (_Float16*)alloc((size_t)BL * DIN * 2);

  cvt_f16_kernel<<<(BL * DM + 255) / 256, 256, 0, stream>>>(x, x_h, BL * DM);
  cvt_f16_kernel<<<(2 * DIN * DM + 255) / 256, 256, 0, stream>>>(w_in, win_h, 2 * DIN * DM);
  cvt_f16_kernel<<<(DM * DIN + 255) / 256, 256, 0, stream>>>(w_out, wout_h, DM * DIN);
  pad_xprojw_kernel<<<(KK * 48 * DIN + 255) / 256, 256, 0, stream>>>(xproj_w, xpw_h);
  pad_dtw_kernel<<<(KK * DIN * 32 + 255) / 256, 256, 0, stream>>>(dt_w, dtw_h);

  inproj_gemm_kernel<<<dim3(BL / 16, 6), 128, 0, stream>>>(x_h, win_h, xc, zb);
  conv_scan_kernel<<<(BB * DIN * LL + 255) / 256, 256, 0, stream>>>(xc, conv_w, conv_b, xsT);
  xproj_gemm_kernel<<<dim3(LL / 16, 1, BB * KK), 96, 0, stream>>>(xsT, xpw_h, dtsR, BCb);
  dt_gemm_kernel<<<dim3(LL / 16, 3, BB * KK), 128, 0, stream>>>(dtsR, dtw_h, dtT);
  scan_kernel<<<(BB * KK * DIN * NS) / 256, 256, 0, stream>>>(xsT, dtT, BCb, A_logs, Ds, dt_b, ysb);
  merge_ln_gate_kernel<<<BL, DIN, 0, stream>>>(ysb, zb, ln_g, ln_b, yg);
  out_gemm_kernel<<<dim3(BL / 16, 2), 96, 0, stream>>>(yg, wout_h, out);
}
